// LCNNConv2d_62723702391662
// MI455X (gfx1250) — matrix operand order repaired and checked by reading, not compiled, not run on hardware
//
#include <hip/hip_runtime.h>
#include <stdint.h>

// CDNA5 WMMA vector types (wave32)
typedef __attribute__((ext_vector_type(16))) __bf16 v16bf;
typedef __attribute__((ext_vector_type(8)))  float  v8f;

__device__ __forceinline__ unsigned short f32_to_bf16_rne(float f) {
    uint32_t u = __builtin_bit_cast(uint32_t, f);
    uint32_t r = u + 0x7fffu + ((u >> 16) & 1u);   // round-to-nearest-even
    return (unsigned short)(r >> 16);
}

// ---------------------------------------------------------------------------
// Kernel 1: global top-k threshold on |lookup_weights| (keep 8192 of 16384).
// |w| >= 0 so its IEEE bit pattern is monotone as uint -> binary search.
// Finds largest u with count(bits(|w|) >= u) >= KEEP.
// ---------------------------------------------------------------------------
__global__ void topk_threshold_kernel(const float* __restrict__ lw,
                                      unsigned* __restrict__ thr_out) {
    const int N = 128 * 128;
    const int KEEP = 8192;
    __shared__ unsigned s_lo, s_hi;
    __shared__ int s_cnt;
    const int tid = threadIdx.x;
    if (tid == 0) { s_lo = 0u; s_hi = 0x7f800000u; }
    __syncthreads();
    for (int it = 0; it < 32; ++it) {
        unsigned lo = s_lo, hi = s_hi;
        if (lo + 1 >= hi) break;                 // uniform across block
        unsigned mid = lo + ((hi - lo) >> 1);
        if (tid == 0) s_cnt = 0;
        __syncthreads();
        int c = 0;
        for (int i = tid; i < N; i += blockDim.x) {
            unsigned a = __builtin_bit_cast(unsigned, lw[i]) & 0x7fffffffu;
            c += (a >= mid) ? 1 : 0;
        }
        atomicAdd(&s_cnt, c);
        __syncthreads();
        if (tid == 0) { if (s_cnt >= KEEP) s_lo = mid; else s_hi = mid; }
        __syncthreads();
    }
    if (tid == 0) *thr_out = s_lo;
}

// ---------------------------------------------------------------------------
// Kernel 2: fuse sparsified lookup_weights with dictionary into effective conv
// weights, emitted directly in WMMA B-fragment layout (bf16 pairs):
//   wp[ks][nt][lane][v]  (uint32 = packed bf16 pair)
//   o = nt*16 + (lane&15)
//   K = ks*32 + (lane>=16 ? 16 : 0) + 2*v + {0,1};  K = tap*64 + c
//   W_eff[o][K] = sum_d mask(lw[o][d]) * dict[d][c][dh][dw]
// ---------------------------------------------------------------------------
__global__ void pack_weights_kernel(const float* __restrict__ lw,
                                    const float* __restrict__ dict,
                                    const unsigned* __restrict__ thr_p,
                                    unsigned* __restrict__ wp) {
    const unsigned thr = *thr_p;
    int idx = blockIdx.x * blockDim.x + threadIdx.x;
    const int TOTAL = 18 * 8 * 32 * 8;
    if (idx >= TOTAL) return;
    int v    = idx & 7;
    int lane = (idx >> 3) & 31;
    int nt   = (idx >> 8) & 7;
    int ks   = idx >> 11;
    int o     = nt * 16 + (lane & 15);
    int kbase = ks * 32 + ((lane >> 4) ? 16 : 0) + 2 * v;
    unsigned packed = 0u;
    for (int p = 0; p < 2; ++p) {
        int k   = kbase + p;       // 0..575
        int tap = k >> 6;          // 0..8
        int c   = k & 63;
        int dh  = tap / 3, dw = tap % 3;
        float acc = 0.f;
        for (int d = 0; d < 128; ++d) {
            float w = lw[o * 128 + d];
            unsigned a = __builtin_bit_cast(unsigned, w) & 0x7fffffffu;
            float wm = (a >= thr) ? w : 0.f;
            acc += wm * dict[(d * 64 + c) * 9 + dh * 3 + dw];
        }
        packed |= (unsigned)f32_to_bf16_rne(acc) << (16 * p);
    }
    wp[idx] = packed;
}

// ---------------------------------------------------------------------------
// Kernel 3: implicit-GEMM 3x3 conv via v_wmma_f32_16x16x32_bf16.
// Block = 256 threads (8 waves). Tile: 8 rows x 16 cols x 128 out-channels.
// Wave (wm, wn): 2 M-tiles (rows) x 4 N-tiles (16-ch groups).
// Double-buffered fragment pipeline: loads for step ks+1 are issued before
// the WMMAs of step ks so global/LDS latency overlaps the matrix pipe.
// ---------------------------------------------------------------------------
struct Frags {
    v16bf A[2];
    v16bf B[4];
};

__device__ __forceinline__ void load_frags(int ks,
                                           const unsigned short* __restrict__ xt,
                                           const uint4* __restrict__ wg,
                                           int wave_m, int wave_n, int lane,
                                           Frags& f) {
    const int tap = ks >> 1;
    const int dh = tap / 3, dw = tap % 3;
    const int laneHi = lane >> 4;
    const int wi = lane & 15;
    const int cb = ((ks & 1) << 5) + (laneHi << 4);

    // B fragments first: longest-latency (global) loads issue earliest.
#pragma unroll
    for (int j = 0; j < 4; ++j) {
        const int nt = wave_n * 4 + j;
        const uint4* bp = wg + (((ks * 8 + nt) * 32 + lane) << 1);
        union { uint4 q[2]; v16bf v; } ub;
        ub.q[0] = bp[0];
        ub.q[1] = bp[1];
        f.B[j] = ub.v;
    }
    // A fragments: 16x32 bf16, two ds_load_b128 each (16B aligned)
#pragma unroll
    for (int mt = 0; mt < 2; ++mt) {
        const int row = wave_m * 2 + mt;
        const unsigned short* base = &xt[((row + dh) * 18 + (wi + dw)) * 64 + cb];
        union { uint4 q[2]; v16bf v; } ua;
        ua.q[0] = *(const uint4*)(base);        // c = cb .. cb+7
        ua.q[1] = *(const uint4*)(base + 8);    // c = cb+8 .. cb+15
        f.A[mt] = ua.v;
    }
}

__global__ __launch_bounds__(256)
void lcnn_conv_kernel(const float* __restrict__ x,
                      const unsigned* __restrict__ wp,
                      const float* __restrict__ bias,
                      float* __restrict__ out) {
    extern __shared__ char smem[];
    unsigned short* xt = (unsigned short*)smem;    // [10][18][64] bf16 (23040 B)
    float* ot = (float*)smem;                      // [128][133] f32, overlaid

    const int b  = blockIdx.z;
    const int h0 = blockIdx.y * 8;
    const int w0 = blockIdx.x * 16;

    const int lane   = threadIdx.x & 31;
    const int waveid = threadIdx.x >> 5;
    const int wave_m = waveid & 3;     // M offset = wave_m*32
    const int wave_n = waveid >> 2;    // N offset = wave_n*64
    const int laneHi = lane >> 4;
    const int wi     = lane & 15;

    // ---- Stage halo tile of x (f32 global -> bf16 LDS, [y][x][c]) ----------
    for (int e = threadIdx.x; e < 64 * 10 * 18; e += 256) {
        int xx = e % 18;
        int t  = e / 18;
        int y  = t % 10;
        int c  = t / 10;
        int gh = h0 + y - 1, gw = w0 + xx - 1;
        float val = 0.f;
        if (gh >= 0 && gh < 112 && gw >= 0 && gw < 112)
            val = x[((b * 64 + c) * 112 + gh) * 112 + gw];
        xt[(y * 18 + xx) * 64 + c] = f32_to_bf16_rne(val);
    }
    __syncthreads();

    // ---- K loop: 18 steps of K=32 (tap-major, channel-minor) --------------
    v8f acc[2][4];
#pragma unroll
    for (int i = 0; i < 2; ++i)
#pragma unroll
        for (int j = 0; j < 4; ++j)
#pragma unroll
            for (int r = 0; r < 8; ++r) acc[i][j][r] = 0.f;

    const uint4* wg = (const uint4*)wp;

    Frags frag[2];
    load_frags(0, xt, wg, wave_m, wave_n, lane, frag[0]);

#pragma unroll
    for (int ks = 0; ks < 18; ++ks) {
        Frags& cur = frag[ks & 1];
        if (ks + 1 < 18)
            load_frags(ks + 1, xt, wg, wave_m, wave_n, lane, frag[(ks + 1) & 1]);
#pragma unroll
        for (int j = 0; j < 4; ++j) {
#pragma unroll
            for (int mt = 0; mt < 2; ++mt) {
                acc[mt][j] = __builtin_amdgcn_wmma_f32_16x16x32_bf16(
                    false, cur.A[mt], false, cur.B[j], (short)0, acc[mt][j],
                    false, false);
            }
        }
    }

    // ---- Epilogue: transpose through LDS for coalesced f32 stores ---------
    __syncthreads();   // done with xt; reuse LDS as ot
    const int PITCH = 133;  // stride 133 % 64 banks = 5 -> conflict-free
#pragma unroll
    for (int mt = 0; mt < 2; ++mt) {
        const int row = wave_m * 2 + mt;
#pragma unroll
        for (int j = 0; j < 4; ++j) {
            const int nt = wave_n * 4 + j;
            const int o  = nt * 16 + wi;
#pragma unroll
            for (int r = 0; r < 8; ++r) {
                int mg = row * 16 + r + laneHi * 8;   // pixel index in tile
                ot[mg * PITCH + o] = acc[mt][j][r];
            }
        }
    }
    __syncthreads();

    // 4-wide (b128) coalesced stores with bias add
    for (int e = threadIdx.x; e < 128 * 32; e += 256) {
        int mq = e & 31;          // group of 4 pixels
        int o  = e >> 5;
        int m0 = mq * 4;
        int h = h0 + (m0 >> 4);
        int w = w0 + (m0 & 15);
        float bo = bias[o];
        float4 val;
        val.x = ot[(m0 + 0) * PITCH + o] + bo;
        val.y = ot[(m0 + 1) * PITCH + o] + bo;
        val.z = ot[(m0 + 2) * PITCH + o] + bo;
        val.w = ot[(m0 + 3) * PITCH + o] + bo;
        *(float4*)&out[((b * 128 + o) * 112 + h) * 112 + w] = val;
    }
}

// ---------------------------------------------------------------------------
extern "C" void kernel_launch(void* const* d_in, const int* in_sizes, int n_in,
                              void* d_out, int out_size, void* d_ws, size_t ws_size,
                              hipStream_t stream) {
    const float* x    = (const float*)d_in[0];   // (32,64,112,112)
    const float* dict = (const float*)d_in[1];   // (128,64,3,3)
    const float* lw   = (const float*)d_in[2];   // (128,128)
    const float* bias = (const float*)d_in[3];   // (128,)
    float* out = (float*)d_out;                  // (32,128,112,112)

    unsigned* thr = (unsigned*)d_ws;                           // 4 B
    unsigned* wp  = (unsigned*)((char*)d_ws + 256);            // 147456 B packed W_eff

    topk_threshold_kernel<<<1, 256, 0, stream>>>(lw, thr);

    const int pack_total = 18 * 8 * 32 * 8;                    // 36864
    pack_weights_kernel<<<(pack_total + 255) / 256, 256, 0, stream>>>(lw, dict, thr, wp);

    dim3 grid(7, 14, 32);                                      // w-tiles, h-tiles, batch
    size_t smem = 128 * 133 * sizeof(float);                   // 68096 B (>= x-tile)
    lcnn_conv_kernel<<<grid, 256, smem, stream>>>(x, wp, bias, out);
    (void)in_sizes; (void)n_in; (void)out_size; (void)ws_size;
}
